// Attention_51539607637
// MI455X (gfx1250) — matrix operand order repaired and checked by reading, not compiled, not run on hardware
//
#include <hip/hip_runtime.h>

typedef unsigned short u16;
typedef unsigned int   u32;
typedef unsigned long long u64;
typedef __attribute__((ext_vector_type(16))) __bf16 v16bf;
typedef __attribute__((ext_vector_type(8)))  float  v8f;

struct U4 { u32 x, y, z, w; };          // POD 16-byte chunk for b128 loads
union Frag { v16bf v; U4 q[2]; u32 u[8]; };

constexpr int   NB      = 2;            // batch
constexpr int   NSEQ    = 2048;
constexpr int   DIMM    = 512;
constexpr int   HEADS   = 8;
constexpr int   DHEAD   = 64;
constexpr int   INNER   = 512;
constexpr float SCALE   = 0.125f;       // 64^-0.5
constexpr int   SSTRIDE = 2052;         // padded LDS row stride (floats)
constexpr int   STAGE_BYTES_PER_WAVE = 4096;   // 2 x 2KB K-tile double buffer

__device__ inline u16 f2bf(float f) {   // round-to-nearest-even fp32 -> bf16 bits
  u32 u = __float_as_uint(f);
  u += 0x7FFFu + ((u >> 16) & 1u);
  return (u16)(u >> 16);
}
__device__ inline u32 pack2(u16 lo, u16 hi) { return (u32)lo | ((u32)hi << 16); }

__device__ inline v8f wmma_bf16(const Frag& a, const Frag& b, v8f c) {
  return __builtin_amdgcn_wmma_f32_16x16x32_bf16(false, a.v, false, b.v,
                                                 (short)0, c, false, false);
}

// -------- async Global -> LDS copy of one 16-row x 64-col bf16 K tile (2 KB) --
// 128 x 16B chunks; each lane copies 4 chunks via global_load_async_to_lds_b128.
__device__ __forceinline__ void async_copy_ktile(const u16* gbase, char* ldsbuf,
                                                 int lane) {
  #pragma unroll
  for (int i = 0; i < 4; ++i) {
    const int chunk = i * 32 + lane;        // 0..127
    const int row = chunk >> 3, sub = chunk & 7;
    const u16* g = gbase + row * 64 + sub * 8;
    const u32 loff = (u32)(uintptr_t)(ldsbuf + row * 128 + sub * 16);
    asm volatile("global_load_async_to_lds_b128 %0, %1, off"
                 :: "v"(loff), "v"((u64)(uintptr_t)g)
                 : "memory");
  }
}
__device__ __forceinline__ void wait_asynccnt4() {
  asm volatile("s_wait_asynccnt 0x4" ::: "memory");
}
__device__ __forceinline__ void wait_asynccnt0() {
  asm volatile("s_wait_asynccnt 0x0" ::: "memory");
}

// ---------------------------------------------------------------- prep kernels
__global__ __launch_bounds__(256) void cvt_bf16_k(const float* __restrict__ src,
                                                  u16* __restrict__ dst, int n) {
  int i = blockIdx.x * 256 + threadIdx.x;
  if (i < n) dst[i] = f2bf(src[i]);
}

// dst[c*rows + r] = bf16(src[r*cols + c])   (stores W^T so GEMM B-frags are contiguous)
__global__ __launch_bounds__(256) void transpose_bf16_k(const float* __restrict__ src,
                                                        u16* __restrict__ dst,
                                                        int rows, int cols) {
  int i = blockIdx.x * 256 + threadIdx.x;
  if (i < rows * cols) {
    int r = i / cols, c = i - r * cols;
    dst[c * rows + r] = f2bf(src[i]);
  }
}

// --------------------------------------------------- kernel 1: qkv = x @ W_qkv
__global__ __launch_bounds__(256) void gemm_qkv_k(const u16* __restrict__ A,
                                                  const u16* __restrict__ Bt,
                                                  u16* __restrict__ Qb,
                                                  u16* __restrict__ Kb,
                                                  u16* __restrict__ Vtb) {
  const int tid = threadIdx.x;
  const int lane = tid & 31, w = tid >> 5;
  const int lane16 = lane & 15, half = lane >> 4;
  const int m0   = blockIdx.x * 16;
  const int col0 = (blockIdx.y * 8 + w) * 16;
  const u16* ar = A  + (m0   + lane16) * DIMM;
  const u16* br = Bt + (col0 + lane16) * DIMM;
  v8f c = {};
  #pragma unroll 4
  for (int k0 = 0; k0 < DIMM; k0 += 32) {
    Frag a, b;
    const int ka = k0 + half * 8;         // A frag: K = ka..ka+7, ka+16..ka+23
    a.q[0] = *(const U4*)(ar + ka);
    a.q[1] = *(const U4*)(ar + ka + 16);
    const int kb = k0 + half * 8;         // B frag: K = kb..kb+7, kb+16..kb+23 of column lane16
    b.q[0] = *(const U4*)(br + kb);
    b.q[1] = *(const U4*)(br + kb + 16);
    __builtin_prefetch(ar + ka + 64, 0, 1);
    c = wmma_bf16(a, b, c);
  }
  const int col = col0 + lane16;
  #pragma unroll
  for (int r = 0; r < 8; ++r) {           // D: M = r + 8*half, N = lane16
    const int rg = m0 + r + half * 8;
    const int b_ = rg >> 11, n = rg & (NSEQ - 1);
    const float val = c[r];
    if (col < INNER) {
      const int h = col >> 6, d = col & 63;
      Qb[((b_ * HEADS + h) * NSEQ + n) * DHEAD + d] = f2bf(val * SCALE);
    } else if (col < 2 * INNER) {
      const int j = col - INNER, h = j >> 6, d = j & 63;
      Kb[((b_ * HEADS + h) * NSEQ + n) * DHEAD + d] = f2bf(val);
    } else {
      const int j = col - 2 * INNER, h = j >> 6, d = j & 63;
      Vtb[((b_ * HEADS + h) * DHEAD + d) * NSEQ + n] = f2bf(val);  // V transposed
    }
  }
}

// -------------------------------- kernel 2: fused sparsemax attention per 16 rows
__global__ __launch_bounds__(256) void attn_k(const u16* __restrict__ Q,
                                              const u16* __restrict__ K,
                                              const u16* __restrict__ Vt,
                                              u16* __restrict__ O) {
  extern __shared__ float smem[];
  float* S    = smem;                 // 16 * SSTRIDE floats
  float* taub = smem + 16 * SSTRIDE;  // 16 floats
  const int tid = threadIdx.x;
  const int lane = tid & 31, w = tid >> 5;
  const int lane16 = lane & 15, half = lane >> 4;
  const int bh = blockIdx.y;
  const int m0 = blockIdx.x * 16;
  const u16* Qb = Q  + (size_t)bh * NSEQ * DHEAD;
  const u16* Kb = K  + (size_t)bh * NSEQ * DHEAD;
  const u16* Vb = Vt + (size_t)bh * DHEAD * NSEQ;
  char* stg = (char*)(smem + 16 * SSTRIDE + 16) + w * STAGE_BYTES_PER_WAVE;

  // --- Phase 1: S = Q*K^T, K tiles streamed Global->LDS with async copies ---
  Frag a0, a1;                         // Q fragments for k0 = 0 and k0 = 32
  {
    const u16* qr = Qb + (m0 + lane16) * DHEAD + half * 8;
    a0.q[0] = *(const U4*)(qr);      a0.q[1] = *(const U4*)(qr + 16);
    a1.q[0] = *(const U4*)(qr + 32); a1.q[1] = *(const U4*)(qr + 48);
  }
  async_copy_ktile(Kb + (w * 16) * DHEAD, stg, lane);   // prefetch first tile
  int buf = 0;
  for (int ct = w; ct < NSEQ / 16; ct += 8) {
    const int ctn = ct + 8;
    if (ctn < NSEQ / 16) {            // stream next tile into other buffer
      async_copy_ktile(Kb + (ctn * 16) * DHEAD, stg + (buf ^ 1) * 2048, lane);
      wait_asynccnt4();               // current tile landed; next may be in flight
    } else {
      wait_asynccnt0();
    }
    const u16* kst = (const u16*)(stg + buf * 2048) + lane16 * 64 + half * 8;
    Frag b0, b1;
    b0.q[0] = *(const U4*)(kst);      b0.q[1] = *(const U4*)(kst + 16);
    b1.q[0] = *(const U4*)(kst + 32); b1.q[1] = *(const U4*)(kst + 48);
    v8f c = {};
    c = wmma_bf16(a0, b0, c);
    c = wmma_bf16(a1, b1, c);
    float* sc = S + ct * 16 + lane16;
    #pragma unroll
    for (int r = 0; r < 8; ++r) sc[(r + half * 8) * SSTRIDE] = c[r];
    buf ^= 1;
  }
  __syncthreads();

  // --- Phase 2: sparsemax tau, one row per half-wave ---
  {
    const int row = 2 * w + half;
    const float* zr = S + row * SSTRIDE;
    float mx = -3.4e38f;
    for (int i = lane16; i < NSEQ; i += 16) mx = fmaxf(mx, zr[i]);
    #pragma unroll
    for (int o = 1; o < 16; o <<= 1) mx = fmaxf(mx, __shfl_xor(mx, o, 32));
    float lo = mx - 1.0f, hi = mx;     // f(lo) >= 1 >= f(hi), f monotone dec.
    for (int it = 0; it < 26; ++it) {
      const float t = 0.5f * (lo + hi);
      float s = 0.0f;
      for (int i = lane16; i < NSEQ; i += 16) s += fmaxf(zr[i] - t, 0.0f);
      #pragma unroll
      for (int o = 1; o < 16; o <<= 1) s += __shfl_xor(s, o, 32);
      if (s >= 1.0f) lo = t; else hi = t;
    }
    const float t = 0.5f * (lo + hi);  // exact refinement on located support
    float cnt = 0.0f, ssum = 0.0f;
    for (int i = lane16; i < NSEQ; i += 16) {
      const float z = zr[i];
      if (z > t) { cnt += 1.0f; ssum += z; }
    }
    #pragma unroll
    for (int o = 1; o < 16; o <<= 1) {
      cnt += __shfl_xor(cnt, o, 32); ssum += __shfl_xor(ssum, o, 32);
    }
    if (lane16 == 0) taub[row] = (cnt > 0.0f) ? (ssum - 1.0f) / cnt : t;
  }
  __syncthreads();

  // --- Phase 3: O = relu(S - tau) @ V, K-split across 8 waves ---
  v8f acc0 = {}, acc1 = {}, acc2 = {}, acc3 = {};
  const int kwb = w * 256;
  const float trow = taub[lane16];
  const float* pr = S + lane16 * SSTRIDE;
  for (int kc = 0; kc < 8; ++kc) {
    Frag pa;                           // A frag built on the fly: relu(z - tau) -> bf16
    const int ka = kwb + kc * 32 + half * 16;
    #pragma unroll
    for (int t2 = 0; t2 < 4; ++t2) {
      pa.u[t2]     = pack2(f2bf(fmaxf(pr[ka + 2 * t2]      - trow, 0.0f)),
                           f2bf(fmaxf(pr[ka + 2 * t2 + 1]  - trow, 0.0f)));
      pa.u[4 + t2] = pack2(f2bf(fmaxf(pr[ka + 8 + 2 * t2] - trow, 0.0f)),
                           f2bf(fmaxf(pr[ka + 9 + 2 * t2] - trow, 0.0f)));
    }
    const int kg = kwb + kc * 32 + half * 16;
    Frag vb;
    const u16* vr = Vb + lane16 * NSEQ + kg;   // V^T rows contiguous in seq
    vb.q[0] = *(const U4*)(vr); vb.q[1] = *(const U4*)(vr + 8);
    acc0 = wmma_bf16(pa, vb, acc0); vr += 16 * NSEQ;
    vb.q[0] = *(const U4*)(vr); vb.q[1] = *(const U4*)(vr + 8);
    acc1 = wmma_bf16(pa, vb, acc1); vr += 16 * NSEQ;
    vb.q[0] = *(const U4*)(vr); vb.q[1] = *(const U4*)(vr + 8);
    acc2 = wmma_bf16(pa, vb, acc2); vr += 16 * NSEQ;
    vb.q[0] = *(const U4*)(vr); vb.q[1] = *(const U4*)(vr + 8);
    acc3 = wmma_bf16(pa, vb, acc3);
  }
  __syncthreads();                     // S no longer needed: reuse for partials
  #pragma unroll
  for (int r = 0; r < 8; ++r) {
    smem[(((w * 4 + 0) * 8 + r) << 5) + lane] = acc0[r];
    smem[(((w * 4 + 1) * 8 + r) << 5) + lane] = acc1[r];
    smem[(((w * 4 + 2) * 8 + r) << 5) + lane] = acc2[r];
    smem[(((w * 4 + 3) * 8 + r) << 5) + lane] = acc3[r];
  }
  __syncthreads();
  {
    const int b_ = bh >> 3, h = bh & 7;
    for (int idx = tid; idx < 16 * DHEAD; idx += 256) {
      const int m = idx >> 6, d = idx & 63;
      const int dt = d >> 4;
      const int ln = (d & 15) + ((m >> 3) << 4);
      const int r  = m & 7;
      float s = 0.0f;
      #pragma unroll
      for (int ww = 0; ww < 8; ++ww) s += smem[(((ww * 4 + dt) * 8 + r) << 5) + ln];
      O[((size_t)(b_ * NSEQ + m0 + m)) * INNER + h * DHEAD + d] = f2bf(s);
    }
  }
}

// ------------------------------------ kernel 3: out = O @ W_out + b_out (fp32)
__global__ __launch_bounds__(256) void gemm_out_k(const u16* __restrict__ A,
                                                  const u16* __restrict__ Bt,
                                                  const float* __restrict__ bias,
                                                  float* __restrict__ out) {
  const int tid = threadIdx.x;
  const int lane = tid & 31, w = tid >> 5;
  const int lane16 = lane & 15, half = lane >> 4;
  const int m0   = blockIdx.x * 16;
  const int col0 = (blockIdx.y * 8 + w) * 16;
  const u16* ar = A  + (m0   + lane16) * INNER;
  const u16* br = Bt + (col0 + lane16) * INNER;
  v8f c = {};
  #pragma unroll 4
  for (int k0 = 0; k0 < INNER; k0 += 32) {
    Frag a, b;
    const int ka = k0 + half * 8;
    a.q[0] = *(const U4*)(ar + ka);
    a.q[1] = *(const U4*)(ar + ka + 16);
    const int kb = k0 + half * 8;
    b.q[0] = *(const U4*)(br + kb);
    b.q[1] = *(const U4*)(br + kb + 16);
    c = wmma_bf16(a, b, c);
  }
  const int col = col0 + lane16;
  const float bv = bias[col];
  #pragma unroll
  for (int r = 0; r < 8; ++r) {
    const int rg = m0 + r + half * 8;
    out[(size_t)rg * DIMM + col] = c[r] + bv;
  }
}

// ------------------------------------------------------------------- launcher
extern "C" void kernel_launch(void* const* d_in, const int* in_sizes, int n_in,
                              void* d_out, int out_size, void* d_ws, size_t ws_size,
                              hipStream_t stream) {
  const float* x    = (const float*)d_in[0];
  const float* Wqkv = (const float*)d_in[1];
  const float* Wout = (const float*)d_in[2];
  const float* bout = (const float*)d_in[3];
  float* out = (float*)d_out;

  unsigned char* p = (unsigned char*)d_ws;
  u16* x_bf   = (u16*)p; p += (size_t)4096 * 512  * 2;
  u16* wqkv_t = (u16*)p; p += (size_t)1536 * 512  * 2;
  u16* wout_t = (u16*)p; p += (size_t)512  * 512  * 2;
  u16* Qb     = (u16*)p; p += (size_t)NB * HEADS * NSEQ * DHEAD * 2;
  u16* Kb     = (u16*)p; p += (size_t)NB * HEADS * NSEQ * DHEAD * 2;
  u16* Vtb    = (u16*)p; p += (size_t)NB * HEADS * NSEQ * DHEAD * 2;
  u16* Ob     = (u16*)p; p += (size_t)4096 * 512 * 2;
  (void)ws_size; (void)n_in; (void)in_sizes; (void)out_size;

  cvt_bf16_k<<<(4096 * 512 + 255) / 256, 256, 0, stream>>>(x, x_bf, 4096 * 512);
  transpose_bf16_k<<<(512 * 1536 + 255) / 256, 256, 0, stream>>>(Wqkv, wqkv_t, 512, 1536);
  transpose_bf16_k<<<(512 * 512  + 255) / 256, 256, 0, stream>>>(Wout, wout_t, 512, 512);

  gemm_qkv_k<<<dim3(4096 / 16, 1536 / 128), 256, 0, stream>>>(x_bf, wqkv_t, Qb, Kb, Vtb);

  const int smem_bytes = (16 * SSTRIDE + 16) * (int)sizeof(float)
                       + 8 * STAGE_BYTES_PER_WAVE;          // ~164 KB < 320 KB WGP LDS
  hipFuncSetAttribute((const void*)attn_k,
                      hipFuncAttributeMaxDynamicSharedMemorySize, smem_bytes);
  attn_k<<<dim3(NSEQ / 16, NB * HEADS), 256, smem_bytes, stream>>>(Qb, Kb, Vtb, Ob);

  gemm_out_k<<<dim3(4096 / 16, 512 / 128), 256, 0, stream>>>(Ob, wout_t, bout, out);
}
